// Model_3496103379307
// MI455X (gfx1250) — matrix operand order repaired and checked by reading, not compiled, not run on hardware
//
#include <hip/hip_runtime.h>
#include <stdint.h>

#define D_MAIN  128
#define D_BLOCK 256
#define N_NUM   64
#define N_CANDS 100000
#define BATCH   1024
#define CTX     96
#define NSPLIT  8
#define NSPC    (N_CANDS / NSPLIT)   /* 12500 */
#define TKBUF   256
#define TKEEP   96
#define NEGF    (-1.0e30f)

typedef __attribute__((ext_vector_type(16))) _Float16 v16h;
typedef __attribute__((ext_vector_type(8)))  _Float16 v8h_t;
typedef __attribute__((ext_vector_type(8)))  float    v8f;

__device__ __forceinline__ v8f wmma16x16x32(v16h a, v16h b, v8f c) {
  // D = A(16x32 f16) * B(32x16 f16) + C(16x16 f32)
  return __builtin_amdgcn_wmma_f32_16x16x32_f16(false, a, false, b, (short)0, c,
                                                false, false);
}

// Async DMA: 16 bytes global -> LDS, tracked by ASYNCcnt (no VGPR return).
__device__ __forceinline__ void async_copy_b128(uint32_t lds_off,
                                                const void* gptr) {
  asm volatile("global_load_async_to_lds_b128 %0, %1, off" ::"v"(lds_off),
               "v"((uint64_t)(uintptr_t)gptr)
               : "memory");
}
__device__ __forceinline__ void wait_asynccnt0() {
  asm volatile("s_wait_asynccnt 0x0" ::: "memory");
}

// Load 16 contiguous halves (16B-aligned) as a v16h, element e <-> offset e.
__device__ __forceinline__ v16h ld16h(const _Float16* p) {
  v8h_t lo = *(const v8h_t*)p;
  v8h_t hi = *(const v8h_t*)(p + 8);
  v16h r;
#pragma unroll
  for (int i = 0; i < 8; i++) { r[i] = lo[i]; r[i + 8] = hi[i]; }
  return r;
}

// A fragment 16x32 f16 from a row-major f16 array (LDS or global).
// Layout: lane<16 holds row M=lane, K = k0..k0+15 (contiguous);
//         lane>=16 holds row M=lane-16, K = k0+16..k0+31.
__device__ __forceinline__ v16h load_a_f16(const _Float16* rows, int ld, int m0,
                                           int k0) {
  int lane = threadIdx.x & 31;
  int khi = lane >> 4;
  const _Float16* p = rows + (size_t)(m0 + (lane & 15)) * ld + k0 + 16 * khi;
  v8h_t lo = *(const v8h_t*)p;
  v8h_t hi = *(const v8h_t*)(p + 8);
  v16h r;
#pragma unroll
  for (int i = 0; i < 8; i++) { r[i] = lo[i]; r[i + 8] = hi[i]; }
  return r;
}

// B fragment 32x16 from pre-converted f16 weights W[N][K] (PyTorch layout):
// column n = lane&15, K-slice k0 + 16*(lane>>4) + e contiguous in W's row n.
__device__ __forceinline__ v16h load_b_wh(const _Float16* W, int K, int n0,
                                          int k0) {
  int lane = threadIdx.x & 31;
  return ld16h(W + (size_t)(n0 + (lane & 15)) * K + k0 + 16 * (lane >> 4));
}

// B fragment 32x16 from a row-major f16 matrix whose ROWS are the B columns
// (i.e. B = M^T), with row-index clamping for partial tiles.
__device__ __forceinline__ v16h load_b_rows_f16(const _Float16* rows, int ld,
                                                int n0, int k0, int nmax) {
  int lane = threadIdx.x & 31;
  int r = n0 + (lane & 15);
  if (r >= nmax) r = nmax - 1;
  return ld16h(rows + (size_t)r * ld + k0 + 16 * (lane >> 4));
}

// Block-level GEMM: C[32,N] = act(A[32,K] @ W^T + bias).  A is LDS f16, W is
// f16 weights in [N,K] layout.  8 waves cover 2 M-tiles x (N/16) N-tiles.
// mode: 0 = f16 store, 1 = f16 relu store, 2 = f32 accumulate, 3 = f32 store,
//       4 = store both f16 and f32.
__device__ __forceinline__ void gemm32(const _Float16* A, int lda,
                                       const _Float16* W, int K, int N,
                                       const float* bias, _Float16* Ch,
                                       int ldc_h, float* Cf, int ldc_f,
                                       int mode) {
  int wave = threadIdx.x >> 5;
  int lane = threadIdx.x & 31;
  int ntiles = N >> 4;
  for (int t = wave; t < 2 * ntiles; t += 8) {
    int mt = t & 1, nt = t >> 1;
    v8f acc = {};
    for (int ks = 0; ks < K; ks += 32) {
      v16h a = load_a_f16(A, lda, mt * 16, ks);
      v16h b = load_b_wh(W, K, nt * 16, ks);
      acc = wmma16x16x32(a, b, acc);
    }
    int col = nt * 16 + (lane & 15);
    float bv = bias ? bias[col] : 0.0f;
#pragma unroll
    for (int i = 0; i < 8; i++) {
      int row = mt * 16 + i + 8 * (lane >> 4);
      float v = acc[i] + bv;
      if (mode == 1) v = v > 0.0f ? v : 0.0f;
      if (mode <= 1 || mode == 4) Ch[(size_t)row * ldc_h + col] = (_Float16)v;
      if (mode == 2) Cf[(size_t)row * ldc_f + col] += v;
      if (mode == 3 || mode == 4) Cf[(size_t)row * ldc_f + col] = v;
    }
  }
}

// Per-row LayerNorm (d=128), thread-serial; caller gates on tid<32.
__device__ __forceinline__ void ln_row(const float* s, _Float16* d,
                                       const float* g, const float* b,
                                       bool relu) {
  float mu = 0.0f;
#pragma unroll
  for (int c = 0; c < D_MAIN; c++) mu += s[c];
  mu *= (1.0f / D_MAIN);
  float var = 0.0f;
#pragma unroll
  for (int c = 0; c < D_MAIN; c++) { float t = s[c] - mu; var += t * t; }
  var *= (1.0f / D_MAIN);
  float rs = rsqrtf(var + 1e-5f);
#pragma unroll
  for (int c = 0; c < D_MAIN; c++) {
    float v = (s[c] - mu) * rs * g[c] + b[c];
    if (relu) v = v > 0.0f ? v : 0.0f;
    d[c] = (_Float16)v;
  }
}

// Cooperative descending bitonic sort of n (power of two) key/idx pairs in LDS.
// Uniform control flow: every thread of the block runs identical loop bounds.
__device__ __forceinline__ void bitonic_desc(float* s, int* ix, int n, int tsub,
                                             int nthr) {
  for (int k = 2; k <= n; k <<= 1) {
    for (int j = k >> 1; j > 0; j >>= 1) {
      __syncthreads();
      for (int i = tsub; i < n; i += nthr) {
        int l = i ^ j;
        if (l > i) {
          bool desc = ((i & k) == 0);
          float a = s[i], b = s[l];
          if (desc ? (a < b) : (a > b)) {
            s[i] = b; s[l] = a;
            int t = ix[i]; ix[i] = ix[l]; ix[l] = t;
          }
        }
      }
    }
  }
  __syncthreads();
}

// ---------------------------------------------------------------------------
// Kernel 0: one-time f32 -> f16 weight conversion into the workspace arena.
// ---------------------------------------------------------------------------
__global__ __launch_bounds__(256) void cvt_f16_kernel(
    const float* __restrict__ src, _Float16* __restrict__ dst, int n) {
  int i = blockIdx.x * 256 + threadIdx.x;
  if (i < n) dst[i] = (_Float16)src[i];
}

// ---------------------------------------------------------------------------
// Kernel 1: shared encoder.  32 rows/block; whole MLP chained through LDS.
// Input tile staged via async global->LDS DMA (ASYNCcnt path).
// ---------------------------------------------------------------------------
__global__ __launch_bounds__(256) void encode_kernel(
    const float* __restrict__ xin, int nrows, const _Float16* __restrict__ W_in,
    const float* __restrict__ b_in, const _Float16* __restrict__ W1,
    const float* __restrict__ b1, const _Float16* __restrict__ W2,
    const float* __restrict__ b2, const float* __restrict__ mg,
    const float* __restrict__ mb, const _Float16* __restrict__ Wk,
    const float* __restrict__ bk, _Float16* __restrict__ k_out,
    float* __restrict__ norm_out, float* __restrict__ x_out) {
  __shared__ __align__(16) float sAf[32][64];  // async-staged f32 input tile
  __shared__ __align__(16) _Float16 sA[32][72];
  __shared__ float sX[32][128];
  __shared__ __align__(16) _Float16 sXh[32][136];
  __shared__ __align__(16) _Float16 sH[32][264];
  int tid = threadIdx.x;
  int row0 = blockIdx.x * 32;

  // nrows is always a multiple of 32 here, so the tile is always full.
  for (int i = tid; i < (32 * N_NUM) / 4; i += 256) {
    int e = i * 4;
    int r = e >> 6, c = e & 63;
    async_copy_b128((uint32_t)(uintptr_t)&sAf[r][c],
                    xin + (size_t)(row0 + r) * N_NUM + c);
  }
  wait_asynccnt0();
  __syncthreads();
  for (int i = tid; i < 32 * N_NUM; i += 256) {
    int r = i >> 6, c = i & 63;
    sA[r][c] = (_Float16)sAf[r][c];
  }
  __syncthreads();
  // x = xnum @ W_in^T + b_in   (keep f32 residual + f16 operand copy)
  gemm32(&sA[0][0], 72, W_in, N_NUM, D_MAIN, b_in, &sXh[0][0], 136, &sX[0][0],
         128, 4);
  __syncthreads();
  // h = relu(x @ W1^T + b1)
  gemm32(&sXh[0][0], 136, W1, D_MAIN, D_BLOCK, b1, &sH[0][0], 264, nullptr, 0, 1);
  __syncthreads();
  // x += h @ W2^T + b2
  gemm32(&sH[0][0], 264, W2, D_BLOCK, D_MAIN, b2, nullptr, 0, &sX[0][0], 128, 2);
  __syncthreads();
  if (x_out) {
    for (int i = tid; i < 32 * D_MAIN; i += 256) {
      int r = i >> 7, c = i & 127, gr = row0 + r;
      if (gr < nrows) x_out[(size_t)gr * D_MAIN + c] = sX[r][c];
    }
  }
  if (tid < 32) ln_row(sX[tid], sXh[tid], mg, mb, false);
  __syncthreads();
  // k = LN(x) @ Wk^T + bk  (f32 into sX)
  gemm32(&sXh[0][0], 136, Wk, D_MAIN, D_MAIN, bk, nullptr, 0, &sX[0][0], 128, 3);
  __syncthreads();
  for (int i = tid; i < 32 * D_MAIN; i += 256) {
    int r = i >> 7, c = i & 127, gr = row0 + r;
    if (gr < nrows) k_out[(size_t)gr * D_MAIN + c] = (_Float16)sX[r][c];
  }
  if (tid < 32) {
    int gr = row0 + tid;
    if (gr < nrows) {
      float s = 0.0f;
#pragma unroll
      for (int c = 0; c < D_MAIN; c++) { float v = sX[tid][c]; s += v * v; }
      norm_out[gr] = s;
    }
  }
}

// ---------------------------------------------------------------------------
// Kernel 2: fused sims-GEMM + streaming top-96 over one candidate split.
// Block = 16 batch rows; 8 waves stream 128 candidates/iteration via WMMA.
// ---------------------------------------------------------------------------
__global__ __launch_bounds__(256) void topk_split_kernel(
    const _Float16* __restrict__ kb, const float* __restrict__ kbnorm,
    const _Float16* __restrict__ ck, const float* __restrict__ cnorm,
    float* __restrict__ osims, int* __restrict__ oidx) {
  __shared__ float bs[16][TKBUF];
  __shared__ int bi[16][TKBUF];
  __shared__ int cnt[16];
  __shared__ float thr[16];
  __shared__ float bn[16];
  __shared__ int needc;

  const int tid = threadIdx.x;
  const int wave = tid >> 5;
  const int lane = tid & 31;
  const int row0 = blockIdx.x * 16;
  const int cbase = blockIdx.y * NSPC;
  const int climit = cbase + NSPC;

  if (tid < 16) { cnt[tid] = 0; thr[tid] = NEGF; bn[tid] = kbnorm[row0 + tid]; }
  __syncthreads();

  v16h afr[4];
#pragma unroll
  for (int s = 0; s < 4; s++)
    afr[s] = load_a_f16(kb + (size_t)row0 * D_MAIN, D_MAIN, 0, 32 * s);

  const int nchunk = (NSPC + 127) / 128;
  for (int ch = 0; ch < nchunk; ch++) {
    int c0 = cbase + ch * 128 + wave * 16;
    int cand = c0 + (lane & 15);
    int candc = cand < climit ? cand : climit - 1;
    if (ch + 1 < nchunk) {  // stream next chunk into cache (global_prefetch)
      int pc = cand + 128;
      if (pc < climit) __builtin_prefetch(ck + (size_t)pc * D_MAIN, 0, 1);
    }
    v8f acc = {};
#pragma unroll
    for (int s = 0; s < 4; s++) {
      v16h bfr = load_b_rows_f16(ck, D_MAIN, c0, 32 * s, climit);
      acc = wmma16x16x32(afr[s], bfr, acc);
    }
    float cn2 = cnorm[candc];
    bool ok = (cand < climit);
#pragma unroll
    for (int i = 0; i < 8; i++) {
      int r = i + 8 * (lane >> 4);
      float s = 2.0f * acc[i] - bn[r] - cn2;
      if (ok && s > thr[r]) {
        int p = atomicAdd(&cnt[r], 1);
        if (p < TKBUF) { bs[r][p] = s; bi[r][p] = cand; }
      }
    }
    __syncthreads();
    if (tid == 0) {
      int f = 0;
      for (int r = 0; r < 16; r++)
        if (cnt[r] > TKBUF - 128) f = 1;  // next chunk adds <=128/row
      needc = f;
    }
    __syncthreads();
    if (needc) {  // compact all rows: sort 256, keep top-96, raise threshold
      for (int pr = 0; pr < 8; pr++) {
        int r = pr * 2 + (tid >> 7);
        int ts = tid & 127;
        int c = cnt[r];
        for (int j = ts; j < TKBUF; j += 128)
          if (j >= c) { bs[r][j] = NEGF; bi[r][j] = -1; }
        __syncthreads();
        bitonic_desc(bs[r], bi[r], TKBUF, ts, 128);
        for (int j = ts; j < TKBUF; j += 128)
          if (j >= TKEEP) { bs[r][j] = NEGF; bi[r][j] = -1; }
        __syncthreads();
        if (ts == 0) {
          cnt[r] = c < TKEEP ? c : TKEEP;
          thr[r] = bs[r][TKEEP - 1];
        }
        __syncthreads();
      }
    }
  }
  // final sort + emit per-split top-96
  for (int pr = 0; pr < 8; pr++) {
    int r = pr * 2 + (tid >> 7);
    int ts = tid & 127;
    int c = cnt[r];
    for (int j = ts; j < TKBUF; j += 128)
      if (j >= c) { bs[r][j] = NEGF; bi[r][j] = -1; }
    __syncthreads();
    bitonic_desc(bs[r], bi[r], TKBUF, ts, 128);
  }
  for (int j = tid; j < 16 * TKEEP; j += 256) {
    int r = j / TKEEP, q = j % TKEEP;
    size_t o = ((size_t)(row0 + r) * NSPLIT + blockIdx.y) * TKEEP + q;
    osims[o] = bs[r][q];
    oidx[o] = bi[r][q];
  }
}

// ---------------------------------------------------------------------------
// Kernel 3: merge 8x96 split results per batch row, global top-96, softmax.
// ---------------------------------------------------------------------------
__global__ __launch_bounds__(256) void merge_softmax_kernel(
    const float* __restrict__ osims, const int* __restrict__ oidx,
    float* __restrict__ probs, int* __restrict__ fidx) {
  __shared__ float ms[1024];
  __shared__ int mi[1024];
  __shared__ float ssum;
  int b = blockIdx.x, tid = threadIdx.x;
  const int TOT = NSPLIT * TKEEP;  // 768
  for (int i = tid; i < 1024; i += 256) {
    if (i < TOT) {
      size_t o = (size_t)b * TOT + i;
      ms[i] = osims[o]; mi[i] = oidx[o];
    } else { ms[i] = NEGF; mi[i] = -1; }
  }
  __syncthreads();
  bitonic_desc(ms, mi, 1024, tid, 256);
  if (tid == 0) {  // serial softmax sum: deterministic order
    float mx = ms[0], s = 0.0f;
    for (int j = 0; j < CTX; j++) { float e = __expf(ms[j] - mx); ms[j] = e; s += e; }
    ssum = s;
  }
  __syncthreads();
  float inv = 1.0f / ssum;
  for (int j = tid; j < CTX; j += 256) {
    probs[(size_t)b * CTX + j] = ms[j] * inv;
    fidx[(size_t)b * CTX + j] = mi[j];
  }
}

// ---------------------------------------------------------------------------
// Kernel 4: T-module over 32 of the 96 context entries of one batch row,
// plus the probability-weighted partial reduction (deterministic, no atomics).
// ---------------------------------------------------------------------------
__global__ __launch_bounds__(256) void context_kernel(
    const _Float16* __restrict__ kb, const _Float16* __restrict__ ck,
    const float* __restrict__ cand_y, const float* __restrict__ w_le,
    const float* __restrict__ b_le, const _Float16* __restrict__ Wt1,
    const float* __restrict__ bt1, const _Float16* __restrict__ Wt2,
    const float* __restrict__ probs, const int* __restrict__ fidx,
    float* __restrict__ ctxpart) {
  __shared__ __align__(16) _Float16 sD[32][136];
  __shared__ __align__(16) _Float16 sH[32][264];
  __shared__ float sV[32][128];
  __shared__ float sp[32];
  __shared__ int si[32];
  int b = blockIdx.x, half = blockIdx.y, tid = threadIdx.x;
  int j0 = half * 32;
  if (tid < 32) {
    size_t ji = (size_t)b * CTX + j0 + tid;
    int ix = fidx[ji];
    si[tid] = ix < 0 ? 0 : ix;
    sp[tid] = probs[ji];
  }
  __syncthreads();
  for (int i = tid; i < 32 * D_MAIN; i += 256) {
    int m = i >> 7, c = i & 127;
    int ci = si[m];
    float kv = (float)kb[(size_t)b * D_MAIN + c];
    float cv = (float)ck[(size_t)ci * D_MAIN + c];
    sD[m][c] = (_Float16)(kv - cv);                 // diff = k - ctx_k
    sV[m][c] = cand_y[ci] * w_le[c] + b_le[c];      // label embedding
  }
  __syncthreads();
  gemm32(&sD[0][0], 136, Wt1, D_MAIN, D_BLOCK, bt1, &sH[0][0], 264, nullptr, 0, 1);
  __syncthreads();
  gemm32(&sH[0][0], 264, Wt2, D_BLOCK, D_MAIN, nullptr, nullptr, 0, &sV[0][0],
         128, 2);
  __syncthreads();
  if (tid < D_MAIN) {
    float a = 0.0f;
#pragma unroll
    for (int m = 0; m < 32; m++) a += sp[m] * sV[m][tid];
    ctxpart[((size_t)half * BATCH + b) * D_MAIN + tid] = a;
  }
}

// ---------------------------------------------------------------------------
// Kernel 5: residual add + prenorm predictor MLP + head.
// ---------------------------------------------------------------------------
__global__ __launch_bounds__(256) void predictor_head_kernel(
    const float* __restrict__ xenc, const float* __restrict__ ctxpart,
    const float* __restrict__ pg, const float* __restrict__ pb,
    const _Float16* __restrict__ Wp1, const float* __restrict__ bp1,
    const _Float16* __restrict__ Wp2, const float* __restrict__ bp2,
    const float* __restrict__ hg, const float* __restrict__ hb,
    const float* __restrict__ Wh, const float* __restrict__ bh,
    float* __restrict__ out) {
  __shared__ float sX[32][128];
  __shared__ __align__(16) _Float16 sXh[32][136];
  __shared__ __align__(16) _Float16 sH[32][264];
  int tid = threadIdx.x;
  int row0 = blockIdx.x * 32;
  for (int i = tid; i < 32 * D_MAIN; i += 256) {
    int r = i >> 7, c = i & 127;
    size_t g = (size_t)(row0 + r) * D_MAIN + c;
    float v = xenc[g];
    v += ctxpart[((size_t)0 * BATCH + row0 + r) * D_MAIN + c];
    v += ctxpart[((size_t)1 * BATCH + row0 + r) * D_MAIN + c];
    v += ctxpart[((size_t)2 * BATCH + row0 + r) * D_MAIN + c];
    sX[r][c] = v;
  }
  __syncthreads();
  if (tid < 32) ln_row(sX[tid], sXh[tid], pg, pb, false);
  __syncthreads();
  gemm32(&sXh[0][0], 136, Wp1, D_MAIN, D_BLOCK, bp1, &sH[0][0], 264, nullptr, 0, 1);
  __syncthreads();
  gemm32(&sH[0][0], 264, Wp2, D_BLOCK, D_MAIN, bp2, nullptr, 0, &sX[0][0], 128, 2);
  __syncthreads();
  if (tid < 32) ln_row(sX[tid], sXh[tid], hg, hb, true);  // relu(LN(x))
  __syncthreads();
  if (tid < 64) {
    int r = tid >> 1, o = tid & 1;
    float s = bh[o];
#pragma unroll
    for (int c = 0; c < D_MAIN; c++) s += (float)sXh[r][c] * Wh[o * D_MAIN + c];
    out[(size_t)(row0 + r) * 2 + o] = s;
  }
}

// ---------------------------------------------------------------------------
extern "C" void kernel_launch(void* const* d_in, const int* in_sizes, int n_in,
                              void* d_out, int out_size, void* d_ws,
                              size_t ws_size, hipStream_t stream) {
  const float* x_num  = (const float*)d_in[0];
  const float* cand_x = (const float*)d_in[1];
  const float* cand_y = (const float*)d_in[2];
  const float* W_in   = (const float*)d_in[3];
  const float* b_in   = (const float*)d_in[4];
  const float* eW1    = (const float*)d_in[5];
  const float* eb1    = (const float*)d_in[6];
  const float* eW2    = (const float*)d_in[7];
  const float* eb2    = (const float*)d_in[8];
  const float* mg     = (const float*)d_in[9];
  const float* mb     = (const float*)d_in[10];
  const float* Wk     = (const float*)d_in[11];
  const float* bk     = (const float*)d_in[12];
  const float* w_le   = (const float*)d_in[13];
  const float* b_le   = (const float*)d_in[14];
  const float* Wt1    = (const float*)d_in[15];
  const float* bt1    = (const float*)d_in[16];
  const float* Wt2    = (const float*)d_in[17];
  const float* pg     = (const float*)d_in[18];
  const float* pb     = (const float*)d_in[19];
  const float* pW1    = (const float*)d_in[20];
  const float* pb1    = (const float*)d_in[21];
  const float* pW2    = (const float*)d_in[22];
  const float* pb2    = (const float*)d_in[23];
  const float* hg     = (const float*)d_in[24];
  const float* hb     = (const float*)d_in[25];
  const float* Wh     = (const float*)d_in[26];
  const float* bh     = (const float*)d_in[27];
  float* out = (float*)d_out;
  (void)in_sizes; (void)n_in; (void)out_size; (void)ws_size;

  uintptr_t base = (uintptr_t)d_ws;
  auto alloc = [&](size_t bytes) -> void* {
    base = (base + 255) & ~(uintptr_t)255;
    void* p = (void*)base;
    base += bytes;
    return p;
  };
  _Float16* ck    = (_Float16*)alloc((size_t)N_CANDS * D_MAIN * 2);
  float*    cn    = (float*)alloc((size_t)N_CANDS * 4);
  _Float16* kb    = (_Float16*)alloc((size_t)BATCH * D_MAIN * 2);
  float*    kbn   = (float*)alloc((size_t)BATCH * 4);
  float*    xenc  = (float*)alloc((size_t)BATCH * D_MAIN * 4);
  float*    osims = (float*)alloc((size_t)BATCH * NSPLIT * TKEEP * 4);
  int*      oidx  = (int*)alloc((size_t)BATCH * NSPLIT * TKEEP * 4);
  float*    probs = (float*)alloc((size_t)BATCH * CTX * 4);
  int*      fidx  = (int*)alloc((size_t)BATCH * CTX * 4);
  float*    ctxp  = (float*)alloc((size_t)3 * BATCH * D_MAIN * 4);
  // f16 weight arena (converted once, L2-resident thereafter)
  _Float16* hW_in = (_Float16*)alloc((size_t)D_MAIN * N_NUM * 2);
  _Float16* hW1   = (_Float16*)alloc((size_t)D_BLOCK * D_MAIN * 2);
  _Float16* hW2   = (_Float16*)alloc((size_t)D_MAIN * D_BLOCK * 2);
  _Float16* hWk   = (_Float16*)alloc((size_t)D_MAIN * D_MAIN * 2);
  _Float16* hWt1  = (_Float16*)alloc((size_t)D_BLOCK * D_MAIN * 2);
  _Float16* hWt2  = (_Float16*)alloc((size_t)D_MAIN * D_BLOCK * 2);
  _Float16* hWp1  = (_Float16*)alloc((size_t)D_BLOCK * D_MAIN * 2);
  _Float16* hWp2  = (_Float16*)alloc((size_t)D_MAIN * D_BLOCK * 2);

  struct { const float* s; _Float16* d; int n; } cv[8] = {
      {W_in, hW_in, D_MAIN * N_NUM},  {eW1, hW1, D_BLOCK * D_MAIN},
      {eW2, hW2, D_MAIN * D_BLOCK},   {Wk, hWk, D_MAIN * D_MAIN},
      {Wt1, hWt1, D_BLOCK * D_MAIN},  {Wt2, hWt2, D_MAIN * D_BLOCK},
      {pW1, hWp1, D_BLOCK * D_MAIN},  {pW2, hWp2, D_MAIN * D_BLOCK}};
  for (int i = 0; i < 8; i++)
    cvt_f16_kernel<<<dim3((cv[i].n + 255) / 256), 256, 0, stream>>>(
        cv[i].s, cv[i].d, cv[i].n);

  encode_kernel<<<dim3(N_CANDS / 32), 256, 0, stream>>>(
      cand_x, N_CANDS, hW_in, b_in, hW1, eb1, hW2, eb2, mg, mb, hWk, bk, ck, cn,
      nullptr);
  encode_kernel<<<dim3(BATCH / 32), 256, 0, stream>>>(
      x_num, BATCH, hW_in, b_in, hW1, eb1, hW2, eb2, mg, mb, hWk, bk, kb, kbn,
      xenc);
  topk_split_kernel<<<dim3(BATCH / 16, NSPLIT), 256, 0, stream>>>(
      kb, kbn, ck, cn, osims, oidx);
  merge_softmax_kernel<<<dim3(BATCH), 256, 0, stream>>>(osims, oidx, probs,
                                                        fidx);
  context_kernel<<<dim3(BATCH, 3), 256, 0, stream>>>(
      kb, ck, cand_y, w_le, b_le, hWt1, bt1, hWt2, probs, fidx, ctxp);
  predictor_head_kernel<<<dim3(BATCH / 32), 256, 0, stream>>>(
      xenc, ctxp, pg, pb, hWp1, pb1, hWp2, pb2, hg, hb, Wh, bh, out);
}
